// MultiHeadAttention_17841294148135
// MI455X (gfx1250) — compile-verified, matrix operand order checked by reading, not run on hardware
//
#include <hip/hip_runtime.h>

#define B_DIM 2
#define T_DIM 2048
#define NH    16
#define DK    64
#define DM    1024   // NH*DK
#define MTOT  4096   // B*T

#define KV_STEP 32
#define KLD 72       // 64 + 8 pad elements -> 36-bank row stride (conflict-free)
#define VLD 40       // 32 + 8 pad elements -> 20-bank row stride (conflict-free)

typedef unsigned short bf16_t;
typedef __attribute__((ext_vector_type(16))) __bf16 bf16x16;
typedef __attribute__((ext_vector_type(8)))  float  floatx8;

static __device__ __forceinline__ bf16_t f32_to_bf16(float f) {
  unsigned u = __float_as_uint(f);
  u += 0x7FFFu + ((u >> 16) & 1u);   // round-to-nearest-even
  return (bf16_t)(u >> 16);
}
static __device__ __forceinline__ unsigned pack2(float a, float b) {
  return (unsigned)f32_to_bf16(a) | ((unsigned)f32_to_bf16(b) << 16);
}

union ABFrag { bf16x16 v; uint4 q[2]; };

// Load one v_wmma_*_16x16x32_bf16 A operand (rows = M) or B operand (matrix
// stored transposed row-major, rows = N) from a row-major bf16 matrix.
// ISA 16-bit layout: lane<16 holds row (lane&15), K chunks {c0..c0+7,
// c0+16..c0+23} with c0 = 8*(lane>>4).
static __device__ __forceinline__ bf16x16 load_ab(const bf16_t* base, int row0, int col0, int ld) {
  int lane = threadIdx.x & 31;
  const bf16_t* p = base + (size_t)(row0 + (lane & 15)) * ld + col0 + ((lane >> 4) << 3);
  ABFrag f;
  f.q[0] = *(const uint4*)(p);
  f.q[1] = *(const uint4*)(p + 16);
  return f.v;
}
// Same pattern out of an LDS tile (base points into a __shared__ array).
static __device__ __forceinline__ bf16x16 load_ab_lds(const bf16_t* base, int row0, int col0, int ld) {
  int lane = threadIdx.x & 31;
  const bf16_t* p = base + (row0 + (lane & 15)) * ld + col0 + ((lane >> 4) << 3);
  ABFrag f;
  f.q[0] = *(const uint4*)(p);
  f.q[1] = *(const uint4*)(p + 16);
  return f.v;
}

static __device__ __forceinline__ floatx8 wmma_bf16(bf16x16 a, bf16x16 b, floatx8 c) {
  return __builtin_amdgcn_wmma_f32_16x16x32_bf16(false, a, false, b, (short)0, c, false, false);
}

// Async global -> LDS copy (16B per lane), tracked by ASYNCcnt. No VGPR data
// round-trip; completion enforced with s_wait_asynccnt.
static __device__ __forceinline__ void async_copy_b128(unsigned lds_off, const bf16_t* gaddr) {
  asm volatile("global_load_async_to_lds_b128 %0, %1, off"
               :: "v"(lds_off), "v"(gaddr)
               : "memory");
}
static __device__ __forceinline__ unsigned lds_off32(const bf16_t* p) {
  return (unsigned)(size_t)p;   // shared aperture keeps LDS offset in addr[31:0]
}

// ---------------- conversion kernels ----------------
__global__ void cvt_bf16_k(const float* __restrict__ s, bf16_t* __restrict__ d, int n) {
  int i = blockIdx.x * blockDim.x + threadIdx.x;
  if (i < n) d[i] = f32_to_bf16(s[i]);
}
__global__ void cvt_tr_bf16_k(const float* __restrict__ s, bf16_t* __restrict__ d, int rows, int cols) {
  int i = blockIdx.x * blockDim.x + threadIdx.x;
  if (i < rows * cols) {
    int r = i / cols, c = i % cols;
    d[(size_t)c * rows + r] = f32_to_bf16(s[i]);
  }
}
__global__ void mask_bias_k(const int* __restrict__ m, float* __restrict__ b, int n) {
  int i = blockIdx.x * blockDim.x + threadIdx.x;
  if (i < n) b[i] = (m[i] != 0) ? 0.0f : -1e30f;
}

// ---------------- generic bf16 WMMA GEMM, 64x64 per wave ----------------
// C[m,n] = sum_k A[m,k] * Bt[n,k].  A: MxK row-major, Bt: NxK row-major.
// MODE 0: bf16 C row-major (ldc)   MODE 1: bf16 C transposed, CT[n*ldc+m]
// MODE 2: f32 C row-major (ldc)
template <int MODE>
__global__ __launch_bounds__(256) void gemm_bf16_k(const bf16_t* __restrict__ A,
                                                   const bf16_t* __restrict__ Bt,
                                                   void* __restrict__ Cp,
                                                   int N, int K, int ldc) {
  int wave = (blockIdx.x << 3) + (threadIdx.x >> 5);
  int lane = threadIdx.x & 31;
  int ntiles = N >> 6;
  int m0 = (wave / ntiles) << 6;
  int n0 = (wave % ntiles) << 6;
  int prow = (lane & 15) + ((lane >> 4) << 4);   // rows 0..31 across lanes

  floatx8 acc[4][4];
  floatx8 z = {};
#pragma unroll
  for (int i = 0; i < 4; ++i)
#pragma unroll
    for (int j = 0; j < 4; ++j) acc[i][j] = z;

  for (int k = 0; k < K; k += 32) {
    // speculative prefetch of the stream 64 columns ahead
    __builtin_prefetch(A  + (size_t)(m0 + prow) * K + k + 64, 0, 1);
    __builtin_prefetch(Bt + (size_t)(n0 + prow) * K + k + 64, 0, 1);
    bf16x16 af[4], bfr[4];
#pragma unroll
    for (int i = 0; i < 4; ++i) af[i] = load_ab(A, m0 + 16 * i, k, K);
#pragma unroll
    for (int j = 0; j < 4; ++j) bfr[j] = load_ab(Bt, n0 + 16 * j, k, K);
#pragma unroll
    for (int i = 0; i < 4; ++i)
#pragma unroll
      for (int j = 0; j < 4; ++j) acc[i][j] = wmma_bf16(af[i], bfr[j], acc[i][j]);
  }

  int half = lane >> 4, nl = lane & 15;
#pragma unroll
  for (int i = 0; i < 4; ++i)
#pragma unroll
    for (int j = 0; j < 4; ++j) {
      if (MODE == 0) {
        bf16_t* C = (bf16_t*)Cp;
        int n = n0 + 16 * j + nl;
#pragma unroll
        for (int r = 0; r < 8; ++r) {
          int m = m0 + 16 * i + 8 * half + r;
          C[(size_t)m * ldc + n] = f32_to_bf16(acc[i][j][r]);
        }
      } else if (MODE == 1) {
        bf16_t* C = (bf16_t*)Cp;
        int n = n0 + 16 * j + nl;
        int mb = m0 + 16 * i + 8 * half;
        uint4 pk = make_uint4(pack2(acc[i][j][0], acc[i][j][1]),
                              pack2(acc[i][j][2], acc[i][j][3]),
                              pack2(acc[i][j][4], acc[i][j][5]),
                              pack2(acc[i][j][6], acc[i][j][7]));
        *(uint4*)&C[(size_t)n * ldc + mb] = pk;
      } else {
        float* C = (float*)Cp;
        int n = n0 + 16 * j + nl;
#pragma unroll
        for (int r = 0; r < 8; ++r) {
          int m = m0 + 16 * i + 8 * half + r;
          C[(size_t)m * ldc + n] = acc[i][j][r];
        }
      }
    }
}

// ---------------- flash attention, async double-buffered K/V in LDS --------
// Block = one (b, h) and 128 q rows: 8 waves x 16-row q tiles sharing the K/V
// stream through LDS. Tile i+1 is fetched with global_load_async_to_lds_b128
// while tile i is consumed (S^T = K.Q^T, O^T = V^T.P^T; P^T repacks in-lane
// into the PV WMMA's B operand).
__global__ __launch_bounds__(256) void attn_k(const bf16_t* __restrict__ Q,
                                              const bf16_t* __restrict__ Kb,
                                              const bf16_t* __restrict__ VT,
                                              const float* __restrict__ mbias,
                                              bf16_t* __restrict__ O) {
  __shared__ bf16_t kt[2][KV_STEP * KLD];   // K rows [kv..kv+32) x 64 d
  __shared__ bf16_t vt[2][DK * VLD];        // V^T rows 64 d x [kv..kv+32)

  int tid = threadIdx.x;
  int lane = tid & 31;
  int bh = blockIdx.x >> 4;            // 32 (b,h) pairs, 16 blocks each
  int h = bh & 15, b = bh >> 4;
  int q0 = ((blockIdx.x & 15) << 7) + ((tid >> 5) << 4);
  int c0 = (lane >> 4) << 3;           // 0 or 8

  const bf16_t* Qh = Q  + (size_t)b * T_DIM * DM + (size_t)h * DK;   // [t, d] ld=DM
  const bf16_t* Kh = Kb + (size_t)b * T_DIM * DM + (size_t)h * DK;   // [t, d] ld=DM
  const bf16_t* Vh = VT + (size_t)h * DK * MTOT + (size_t)b * T_DIM; // [d, t] ld=MTOT
  const float*  mb = mbias + b * T_DIM;

  // async-fill addressing: 256 threads each own one 16B chunk per tile
  int krow = tid >> 3, kcol = (tid & 7) << 3;   // 32 rows x 8 chunks
  int vrow = tid >> 2, vcol = (tid & 3) << 3;   // 64 rows x 4 chunks
  const bf16_t* kgp = Kh + (size_t)krow * DM + kcol;
  const bf16_t* vgp = Vh + (size_t)vrow * MTOT + vcol;
  unsigned koff[2] = { lds_off32(&kt[0][krow * KLD + kcol]),
                       lds_off32(&kt[1][krow * KLD + kcol]) };
  unsigned voff[2] = { lds_off32(&vt[0][vrow * VLD + vcol]),
                       lds_off32(&vt[1][vrow * VLD + vcol]) };

  // Q as B operands for S^T (lane = q column), resident for the whole pass
  bf16x16 bq0 = load_ab(Qh, q0, 0, DM);
  bf16x16 bq1 = load_ab(Qh, q0, 32, DM);

  floatx8 o[4];
  floatx8 z = {};
#pragma unroll
  for (int dt = 0; dt < 4; ++dt) o[dt] = z;
  float m_run = -1e30f, l_run = 0.0f;

  const int NT = T_DIM / KV_STEP;  // 64 tiles
  // prologue: start tile 0 transfers
  async_copy_b128(koff[0], kgp);
  async_copy_b128(voff[0], vgp);

  for (int i = 0; i < NT; ++i) {
    int cur = i & 1;
    int kv = i * KV_STEP;
    if (i + 1 < NT) {
      // launch tile i+1 into the other buffer (released by previous barrier)
      async_copy_b128(koff[cur ^ 1], kgp + (size_t)(kv + KV_STEP) * DM);
      async_copy_b128(voff[cur ^ 1], vgp + (kv + KV_STEP));
      asm volatile("s_wait_asynccnt 0x2" ::: "memory");  // tile i landed (per wave)
    } else {
      asm volatile("s_wait_asynccnt 0x0" ::: "memory");
    }
    __syncthreads();   // tile i landed for all waves

    const bf16_t* ktc = kt[cur];
    const bf16_t* vtc = vt[cur];
    bf16x16 ka0 = load_ab_lds(ktc, 0,  0,  KLD);
    bf16x16 ka1 = load_ab_lds(ktc, 0,  32, KLD);
    bf16x16 kc0 = load_ab_lds(ktc, 16, 0,  KLD);
    bf16x16 kc1 = load_ab_lds(ktc, 16, 32, KLD);
    floatx8 s0 = wmma_bf16(ka1, bq1, wmma_bf16(ka0, bq0, z)); // keys kv..kv+15
    floatx8 s1 = wmma_bf16(kc1, bq1, wmma_bf16(kc0, bq0, z)); // keys kv+16..kv+31

    const float* mp = mb + kv + c0;
    floatx8 mb0 = *(const floatx8*)(mp);        // keys kv+c0..+7
    floatx8 mb1 = *(const floatx8*)(mp + 16);   // keys kv+16+c0..+7
#pragma unroll
    for (int r = 0; r < 8; ++r) {
      s0[r] = s0[r] * 0.125f + mb0[r];          // 1/sqrt(64)
      s1[r] = s1[r] * 0.125f + mb1[r];
    }

    float mx = s0[0];
#pragma unroll
    for (int r = 1; r < 8; ++r) mx = fmaxf(mx, s0[r]);
#pragma unroll
    for (int r = 0; r < 8; ++r) mx = fmaxf(mx, s1[r]);
    mx = fmaxf(mx, __shfl_xor(mx, 16, 32));     // combine key halves (same q)
    float m_new = fmaxf(m_run, mx);
    float alpha = __expf(m_run - m_new);

    float e0[8], e1[8];
    float rs = 0.0f;
#pragma unroll
    for (int r = 0; r < 8; ++r) {
      e0[r] = __expf(s0[r] - m_new);
      e1[r] = __expf(s1[r] - m_new);
      rs += e0[r] + e1[r];
    }
    rs += __shfl_xor(rs, 16, 32);
    l_run = l_run * alpha + rs;
    m_run = m_new;

    // P^T as B operand: VGPRs 0-3 <- keys kv+c0.., VGPRs 4-7 <- keys kv+16+c0..
    ABFrag p;
    p.q[0] = make_uint4(pack2(e0[0], e0[1]), pack2(e0[2], e0[3]),
                        pack2(e0[4], e0[5]), pack2(e0[6], e0[7]));
    p.q[1] = make_uint4(pack2(e1[0], e1[1]), pack2(e1[2], e1[3]),
                        pack2(e1[4], e1[5]), pack2(e1[6], e1[7]));

#pragma unroll
    for (int dt = 0; dt < 4; ++dt) {
      o[dt] = o[dt] * alpha;
      bf16x16 va = load_ab_lds(vtc, dt * 16, 0, VLD);   // V^T rows = d
      o[dt] = wmma_bf16(va, p.v, o[dt]);
    }
    __syncthreads();   // all waves done with buffer `cur` before it is refilled
  }

  float inv = 1.0f / l_run;
  int q = q0 + (lane & 15);
  bf16_t* orow = O + (size_t)(b * T_DIM + q) * DM + h * DK + c0;
#pragma unroll
  for (int dt = 0; dt < 4; ++dt) {
    uint4 pk = make_uint4(pack2(o[dt][0] * inv, o[dt][1] * inv),
                          pack2(o[dt][2] * inv, o[dt][3] * inv),
                          pack2(o[dt][4] * inv, o[dt][5] * inv),
                          pack2(o[dt][6] * inv, o[dt][7] * inv));
    *(uint4*)(orow + dt * 16) = pk;   // stores O[q, h*64 + dt*16 + c0 ..]
  }
}

// ---------------- host side ----------------
extern "C" void kernel_launch(void* const* d_in, const int* in_sizes, int n_in,
                              void* d_out, int out_size, void* d_ws, size_t ws_size,
                              hipStream_t stream) {
  const float* x    = (const float*)d_in[0];
  const int*   mask = (const int*)d_in[1];
  const float* Wq   = (const float*)d_in[2];
  const float* Wk   = (const float*)d_in[3];
  const float* Wv   = (const float*)d_in[4];
  const float* Wo   = (const float*)d_in[5];
  float* out = (float*)d_out;

  char* ws = (char*)d_ws;
  size_t off = 0;
  auto alloc = [&](size_t bytes) {
    void* p = ws + off;
    off += (bytes + 255) & ~(size_t)255;
    return p;
  };
  bf16_t* xb   = (bf16_t*)alloc((size_t)MTOT * DM * 2);
  bf16_t* WqT  = (bf16_t*)alloc((size_t)DM * DM * 2);
  bf16_t* WkT  = (bf16_t*)alloc((size_t)DM * DM * 2);
  bf16_t* WvT  = (bf16_t*)alloc((size_t)DM * DM * 2);
  bf16_t* WoT  = (bf16_t*)alloc((size_t)DM * DM * 2);
  bf16_t* Qb   = (bf16_t*)alloc((size_t)MTOT * DM * 2);
  bf16_t* Kb   = (bf16_t*)alloc((size_t)MTOT * DM * 2);
  bf16_t* VTb  = (bf16_t*)alloc((size_t)DM * MTOT * 2);
  bf16_t* Ob   = (bf16_t*)alloc((size_t)MTOT * DM * 2);
  float*  mbia = (float*)alloc((size_t)MTOT * 4);

  // 1) precision conversion + weight transpose + mask bias
  cvt_bf16_k<<<(MTOT * DM) / 256, 256, 0, stream>>>(x, xb, MTOT * DM);
  cvt_tr_bf16_k<<<(DM * DM) / 256, 256, 0, stream>>>(Wq, WqT, DM, DM);
  cvt_tr_bf16_k<<<(DM * DM) / 256, 256, 0, stream>>>(Wk, WkT, DM, DM);
  cvt_tr_bf16_k<<<(DM * DM) / 256, 256, 0, stream>>>(Wv, WvT, DM, DM);
  cvt_tr_bf16_k<<<(DM * DM) / 256, 256, 0, stream>>>(Wo, WoT, DM, DM);
  mask_bias_k<<<(B_DIM * T_DIM) / 256, 256, 0, stream>>>(mask, mbia, B_DIM * T_DIM);

  // 2) projections: 4096x1024x1024, (M/64)*(N/64)=1024 wave tiles -> 128 blocks
  gemm_bf16_k<0><<<128, 256, 0, stream>>>(xb, WqT, Qb,  DM, DM, DM);
  gemm_bf16_k<0><<<128, 256, 0, stream>>>(xb, WkT, Kb,  DM, DM, DM);
  gemm_bf16_k<1><<<128, 256, 0, stream>>>(xb, WvT, VTb, DM, DM, MTOT); // V stored transposed

  // 3) attention: 32 (b,h) x 16 q-blocks -> 512 blocks, 8 waves each
  attn_k<<<512, 256, 0, stream>>>(Qb, Kb, VTb, mbia, Ob);

  // 4) output projection with f32 result
  gemm_bf16_k<2><<<128, 256, 0, stream>>>(Ob, WoT, out, DM, DM, DM);
}
